// HaloNetT_44684839748315
// MI455X (gfx1250) — matrix operand order repaired and checked by reading, not compiled, not run on hardware
//
#include <hip/hip_runtime.h>
#include <math.h>

// ---------------- types / fragment helpers (CDNA5 WMMA, wave32) ----------------
typedef __attribute__((ext_vector_type(16))) _Float16 v16h;
typedef __attribute__((ext_vector_type(8)))  _Float16 v8h;
typedef __attribute__((ext_vector_type(8)))  float    v8f;

constexpr int BLK   = 8;
constexpr int HALOR = 4;
constexpr int WSZ   = 16;      // BLK + 2*HALOR
constexpr int HEADS = 4;
constexpr int DHD   = 32;
constexpr int CCH   = 128;     // channels == inner dim
constexpr int IMH   = 192;
constexpr int IMW   = 192;
constexpr int NIMG  = 2;
constexpr int HWSZ  = IMH * IMW;          // 36864
constexpr int NPIX  = NIMG * HWSZ;        // 73728
constexpr int NBH   = IMH / BLK;          // 24
constexpr int NBW   = IMW / BLK;          // 24
constexpr int NBLK  = NBH * NBW;          // 576
constexpr int LDP   = 264;                // padded P row (halves)
constexpr int LDV   = 264;                // padded Vt row (halves)

__device__ inline v8f wmma32(v16h a, v16h b, v8f c) {
  // v_wmma_f32_16x16x32_f16  D = A*B + C
  return __builtin_amdgcn_wmma_f32_16x16x32_f16(false, a, false, b, (short)0, c,
                                                false, false);
}
__device__ inline v8f zacc() {
  v8f z;
#pragma unroll
  for (int i = 0; i < 8; ++i) z[i] = 0.0f;
  return z;
}
__device__ inline v16h zfrag() {
  v16h z; _Float16* p = (_Float16*)&z;
#pragma unroll
  for (int i = 0; i < 16; ++i) p[i] = (_Float16)0.0f;
  return z;
}
// A fragment (16x32 f16, MxK): lane covers row M; halves0-7 = K (hi?8..15:0..7),
// halves8-15 = K+16.  `row` points at &A[m][k0] of a row-major [M][K] matrix.
__device__ inline v16h frag_row_a(const _Float16* row, int hi) {
  v16h f;
  ((v8h*)&f)[0] = *(const v8h*)(row + hi * 8);
  ((v8h*)&f)[1] = *(const v8h*)(row + hi * 8 + 16);
  return f;
}
// B fragment (32x16 f16, KxN): lane covers column N; halves0-7 = K (hi?8..15:0..7),
// halves8-15 = K+16.  `row` points at &Bt[n][k0] of a row-major [N][K] matrix.
__device__ inline v16h frag_row_b(const _Float16* row, int hi) {
  v16h f;
  ((v8h*)&f)[0] = *(const v8h*)(row + hi * 8);
  ((v8h*)&f)[1] = *(const v8h*)(row + hi * 8 + 16);
  return f;
}

// ---------------- kernel 0: weight/BN preparation ----------------
__global__ __launch_bounds__(256) void prep_kernel(
    const float* __restrict__ conv1_w, const float* __restrict__ bn1_g,
    const float* __restrict__ bn1_b, const float* __restrict__ bn1_m,
    const float* __restrict__ bn1_v, const float* __restrict__ to_q,
    const float* __restrict__ to_kv, const float* __restrict__ to_out_w,
    const float* __restrict__ rel_h, const float* __restrict__ rel_w,
    const float* __restrict__ conv3_w, const float* __restrict__ bn3_g,
    const float* __restrict__ bn3_b, const float* __restrict__ bn3_m,
    const float* __restrict__ bn3_v,
    _Float16* __restrict__ w1h, _Float16* __restrict__ tqh,
    _Float16* __restrict__ tkh, _Float16* __restrict__ tvh,
    _Float16* __restrict__ toh, _Float16* __restrict__ relwh,
    _Float16* __restrict__ relhh, _Float16* __restrict__ w3h,
    float* __restrict__ s1, float* __restrict__ b1,
    float* __restrict__ s3, float* __restrict__ b3) {
  int tid = blockIdx.x * blockDim.x + threadIdx.x;
  int nth = gridDim.x * blockDim.x;
  // effective conv1 weight (ConvT 1x1): y[o] = sum_i conv1_w[i][o] x[i]
  for (int idx = tid; idx < CCH * CCH; idx += nth) {
    int o = idx / CCH, i = idx % CCH;
    w1h[idx] = (_Float16)conv1_w[i * CCH + o];
    int j = o, c = i;  // [n][k] layouts for B-fragments / A=weights
    tqh[idx] = (_Float16)to_q[c * CCH + j];
    tkh[idx] = (_Float16)to_kv[c * (2 * CCH) + j];
    tvh[idx] = (_Float16)to_kv[c * (2 * CCH) + CCH + j];
    toh[idx] = (_Float16)to_out_w[c * CCH + j];
  }
  // effective conv3 weight: w[t=dy*3+dx][o][i] = conv3_w[i][o][2-dy][2-dx]
  for (int idx = tid; idx < 9 * CCH * CCH; idx += nth) {
    int t = idx / (CCH * CCH);
    int rem = idx % (CCH * CCH);
    int o = rem / CCH, i = rem % CCH;
    int dy = t / 3, dx = t % 3;
    w3h[idx] = (_Float16)conv3_w[((i * CCH + o) * 3 + (2 - dy)) * 3 + (2 - dx)];
  }
  // rel tables padded 31 -> 32 rows (row 31 zero)
  for (int idx = tid; idx < 32 * 32; idx += nth) {
    int r = idx / 32, d = idx % 32;
    relwh[idx] = (r < 31) ? (_Float16)rel_w[r * 32 + d] : (_Float16)0.0f;
    relhh[idx] = (r < 31) ? (_Float16)rel_h[r * 32 + d] : (_Float16)0.0f;
  }
  // folded BN
  for (int c = tid; c < CCH; c += nth) {
    float sv = bn1_g[c] * rsqrtf(bn1_v[c] + 1e-5f);
    s1[c] = sv; b1[c] = bn1_b[c] - bn1_m[c] * sv;
    float sw = bn3_g[c] * rsqrtf(bn3_v[c] + 1e-5f);
    s3[c] = sw; b3[c] = bn3_b[c] - bn3_m[c] * sw;
  }
}

// ---------------- kernel 1: x NCHW f32 -> NHWC f16 ----------------
__global__ __launch_bounds__(256) void nchw_to_nhwc_f16(
    const float* __restrict__ x, _Float16* __restrict__ xh) {
  size_t idx = (size_t)blockIdx.x * blockDim.x + threadIdx.x;
  if (idx >= (size_t)NPIX * CCH) return;
  int n = (int)(idx / ((size_t)CCH * HWSZ));
  size_t r = idx % ((size_t)CCH * HWSZ);
  int c = (int)(r / HWSZ);
  int p = (int)(r % HWSZ);
  xh[((size_t)n * HWSZ + p) * CCH + c] = (_Float16)x[idx];
}

// ---------------- kernel 2: conv1x1 + BN + ReLU  (GEMM, WMMA) ----------------
__global__ __launch_bounds__(256) void conv1x1_bn_relu(
    const _Float16* __restrict__ xh, const _Float16* __restrict__ w1h,
    const float* __restrict__ s1, const float* __restrict__ b1,
    _Float16* __restrict__ y1) {
  int wave = blockIdx.x * 8 + (threadIdx.x >> 5);
  int lane = threadIdx.x & 31, hi = lane >> 4, ln = lane & 15;
  int ct = wave & 7;          // cout tile
  int pt = wave >> 3;         // pixel tile (16 pixels, never straddles a row)
  int co0 = ct * 16;
  size_t p0 = (size_t)pt * 16;
  const _Float16* arow = w1h + (size_t)(co0 + ln) * CCH;  // A: [cout][cin]
  const _Float16* brow = xh + (p0 + ln) * CCH;            // B: [pix][cin]
  v8f acc = zacc();
#pragma unroll
  for (int k0 = 0; k0 < CCH; k0 += 32)
    acc = wmma32(frag_row_a(arow + k0, hi), frag_row_b(brow + k0, hi), acc);
  v8h hv;
#pragma unroll
  for (int r = 0; r < 8; ++r) {
    int co = co0 + hi * 8 + r;
    float v = acc[r] * s1[co] + b1[co];
    hv[r] = (_Float16)(v > 0.f ? v : 0.f);
  }
  *(v8h*)(y1 + (p0 + ln) * CCH + co0 + hi * 8) = hv;  // NHWC, cout-contiguous
}

// ---------------- kernel 3: halo attention ----------------
__global__ __launch_bounds__(256) void halo_attn(
    const _Float16* __restrict__ y1, const _Float16* __restrict__ tqh,
    const _Float16* __restrict__ tkh, const _Float16* __restrict__ tvh,
    const _Float16* __restrict__ toh, const _Float16* __restrict__ relwh,
    const _Float16* __restrict__ relhh, const float* __restrict__ outb,
    _Float16* __restrict__ y2) {
  extern __shared__ char smem[];
  _Float16* Qs = (_Float16*)smem;          // [64][128]   scaled Q, A-layout
  _Float16* Ks = Qs + 64 * CCH;            // [256][128]  K,   B-layout for sim
  _Float16* Vt = Ks + 256 * CCH;           // [128][LDV]  V^T, B-layout for P@V
  _Float16* Ps = Vt + CCH * LDV;           // 8 waves x [16][LDP] probs
  float* QWs = (float*)(Ps + 8 * 16 * LDP);  // 8 x [16][32] rel-w logits
  float* QHs = QWs + 8 * 16 * 32;            // 8 x [16][32] rel-h logits
  _Float16* OutS = (_Float16*)(QHs + 8 * 16 * 32);  // [64][128] head-merged out

  int wv = threadIdx.x >> 5;
  int lane = threadIdx.x & 31, hi = lane >> 4, ln = lane & 15;
  int n = blockIdx.x / NBLK;
  int blk = blockIdx.x % NBLK;
  int by = blk / NBW, bx = blk % NBW;      // by vertical, bx horizontal
  size_t imgbase = (size_t)n * HWSZ;
  const float scale = 0.17677669529663687f;  // 32^-0.5

  // ---- Phase 1+2: Q (32 tiles) / K (128) / V (128) projections ----
  for (int t = wv; t < 288; t += 8) {
    v8f acc = zacc();
    if (t < 32) {                       // Q: A=to_q^T (M=d), B=block rows (N=q)
      int dt = t >> 2, qt = t & 3;
      int q = qt * 16 + ln;
      int gy = by * BLK + (q >> 3), gx = bx * BLK + (q & 7);
      const _Float16* brow = y1 + (imgbase + (size_t)gy * IMW + gx) * CCH;
      const _Float16* arow = tqh + (size_t)(dt * 16 + ln) * CCH;
#pragma unroll
      for (int k0 = 0; k0 < CCH; k0 += 32)
        acc = wmma32(frag_row_a(arow + k0, hi), frag_row_b(brow + k0, hi), acc);
      v8h hv;
#pragma unroll
      for (int r = 0; r < 8; ++r) hv[r] = (_Float16)(acc[r] * scale);
      *(v8h*)(Qs + q * CCH + dt * 16 + hi * 8) = hv;
    } else if (t < 160) {               // K: A=to_k^T (M=d), B=window rows (N=key)
      int t2 = t - 32; int dt = t2 >> 4, kt = t2 & 15;
      int key = kt * 16 + ln;
      int gy = by * BLK - HALOR + (key >> 4);
      int gx = bx * BLK - HALOR + (key & 15);
      bool valid = (gy >= 0 && gy < IMH && gx >= 0 && gx < IMW);
      const _Float16* brow = y1 + (imgbase + (long)gy * IMW + gx) * CCH;
      const _Float16* arow = tkh + (size_t)(dt * 16 + ln) * CCH;
#pragma unroll
      for (int k0 = 0; k0 < CCH; k0 += 32) {
        v16h b = valid ? frag_row_b(brow + k0, hi) : zfrag();
        acc = wmma32(frag_row_a(arow + k0, hi), b, acc);
      }
      v8h hv;
#pragma unroll
      for (int r = 0; r < 8; ++r) hv[r] = (_Float16)acc[r];
      *(v8h*)(Ks + key * CCH + dt * 16 + hi * 8) = hv;
    } else {                            // V: A=window rows (M=key), B=to_v^T (N=d)
      int t3 = t - 160; int dt = t3 >> 4, kt = t3 & 15;
      int key = kt * 16 + ln;
      int gy = by * BLK - HALOR + (key >> 4);
      int gx = bx * BLK - HALOR + (key & 15);
      bool valid = (gy >= 0 && gy < IMH && gx >= 0 && gx < IMW);
      const _Float16* arow = y1 + (imgbase + (long)gy * IMW + gx) * CCH;
      const _Float16* brow = tvh + (size_t)(dt * 16 + ln) * CCH;
#pragma unroll
      for (int k0 = 0; k0 < CCH; k0 += 32) {
        v16h a = valid ? frag_row_a(arow + k0, hi) : zfrag();
        acc = wmma32(a, frag_row_b(brow + k0, hi), acc);
      }
      v8h hv;   // D(m=key,n=d) -> Vt[d][key], key-contiguous store
#pragma unroll
      for (int r = 0; r < 8; ++r) hv[r] = (_Float16)acc[r];
      *(v8h*)(Vt + (dt * 16 + ln) * LDV + kt * 16 + hi * 8) = hv;
    }
  }
  __syncthreads();

  // ---- Phase 3: per (head, q-tile) unit: rel logits, sim, softmax, P@V ----
  _Float16* Pw = Ps + wv * 16 * LDP;
  float* QWw = QWs + wv * 16 * 32;
  float* QHw = QHs + wv * 16 * 32;
  for (int u = wv; u < 16; u += 8) {
    int hd = u >> 2, qt = u & 3;
    v16h aQ = frag_row_a(Qs + (qt * 16 + ln) * CCH + hd * DHD, hi);
    v8f cz = zacc();
    // rel logits: QW = Q_h @ rel_w^T, QH = Q_h @ rel_h^T  (31 padded to 32)
#pragma unroll
    for (int rt = 0; rt < 2; ++rt) {
      v8f dw = wmma32(aQ, frag_row_b(relwh + (rt * 16 + ln) * 32, hi), cz);
      v8f dh = wmma32(aQ, frag_row_b(relhh + (rt * 16 + ln) * 32, hi), cz);
#pragma unroll
      for (int r = 0; r < 8; ++r) {
        QWw[(hi * 8 + r) * 32 + rt * 16 + ln] = dw[r];
        QHw[(hi * 8 + r) * 32 + rt * 16 + ln] = dh[r];
      }
    }
    // sim strip [16 q rows x 256 keys]
    v8f sim[16];
#pragma unroll
    for (int kt = 0; kt < 16; ++kt)
      sim[kt] = wmma32(aQ, frag_row_b(Ks + (kt * 16 + ln) * CCH + hd * DHD, hi), cz);
    // + rel-pos, then mask padded halo keys (key i=kt vertical, j=ln horizontal)
#pragma unroll
    for (int kt = 0; kt < 16; ++kt) {
      int gy = by * BLK - HALOR + kt;
      int gx = bx * BLK - HALOR + ln;
      bool valid = (gy >= 0 && gy < IMH && gx >= 0 && gx < IMW);
#pragma unroll
      for (int r = 0; r < 8; ++r) {
        int q = qt * 16 + hi * 8 + r;
        int qx = q >> 3, qy = q & 7;   // qx vertical, qy horizontal
        float v = sim[kt][r] + QWw[(hi * 8 + r) * 32 + (ln - qy + 15)]
                             + QHw[(hi * 8 + r) * 32 + (kt - qx + 15)];
        sim[kt][r] = valid ? v : -3.0e38f;
      }
    }
    // softmax over 256 keys: reduce 16 reg values then 16 lanes (xor<16 keeps group)
#pragma unroll
    for (int r = 0; r < 8; ++r) {
      float m = -3.4e38f;
#pragma unroll
      for (int kt = 0; kt < 16; ++kt) m = fmaxf(m, sim[kt][r]);
      m = fmaxf(m, __shfl_xor(m, 1)); m = fmaxf(m, __shfl_xor(m, 2));
      m = fmaxf(m, __shfl_xor(m, 4)); m = fmaxf(m, __shfl_xor(m, 8));
      float s = 0.f;
#pragma unroll
      for (int kt = 0; kt < 16; ++kt) {
        float e = __expf(sim[kt][r] - m); sim[kt][r] = e; s += e;
      }
      s += __shfl_xor(s, 1); s += __shfl_xor(s, 2);
      s += __shfl_xor(s, 4); s += __shfl_xor(s, 8);
      float inv = 1.f / s;
#pragma unroll
      for (int kt = 0; kt < 16; ++kt)
        Pw[(hi * 8 + r) * LDP + kt * 16 + ln] = (_Float16)(sim[kt][r] * inv);
    }
    // out_h = P @ V_h   (A = P rows [q][key], B = Vt rows [d][key])
#pragma unroll
    for (int dt2 = 0; dt2 < 2; ++dt2) {
      v8f oacc = zacc();
#pragma unroll
      for (int kc = 0; kc < 8; ++kc) {
        v16h a = frag_row_a(Pw + ln * LDP + kc * 32, hi);
        v16h b = frag_row_b(Vt + (hd * DHD + dt2 * 16 + ln) * LDV + kc * 32, hi);
        oacc = wmma32(a, b, oacc);
      }
#pragma unroll
      for (int r = 0; r < 8; ++r)
        OutS[(qt * 16 + hi * 8 + r) * CCH + hd * DHD + dt2 * 16 + ln] =
            (_Float16)oacc[r];
    }
  }
  __syncthreads();

  // ---- Phase 4: out = OutS @ to_out_w + b, ReLU -> y2 (NHWC f16) ----
  for (int t = wv; t < 32; t += 8) {
    int ct = t & 7, qt = t >> 3;
    int co0 = ct * 16;
    const _Float16* arow = toh + (size_t)(co0 + ln) * CCH;   // A: [o][c]
    const _Float16* brow = OutS + (qt * 16 + ln) * CCH;      // B: [q][c]
    v8f acc = zacc();
#pragma unroll
    for (int k0 = 0; k0 < CCH; k0 += 32)
      acc = wmma32(frag_row_a(arow + k0, hi), frag_row_b(brow + k0, hi), acc);
    int q = qt * 16 + ln;
    size_t gp = imgbase + (size_t)(by * BLK + (q >> 3)) * IMW + bx * BLK + (q & 7);
    v8h hv;
#pragma unroll
    for (int r = 0; r < 8; ++r) {
      float v = acc[r] + outb[co0 + hi * 8 + r];
      hv[r] = (_Float16)(v > 0.f ? v : 0.f);   // relu(halo_attn(...))
    }
    *(v8h*)(y2 + gp * CCH + co0 + hi * 8) = hv;
  }
}

// ---------------- kernel 4: conv3x3 implicit GEMM + BN + residual + ReLU ------
__global__ __launch_bounds__(256) void conv3x3_bn_add_relu(
    const _Float16* __restrict__ y2, const _Float16* __restrict__ w3h,
    const float* __restrict__ s3, const float* __restrict__ b3,
    const float* __restrict__ xin, float* __restrict__ out) {
  int wave = blockIdx.x * 8 + (threadIdx.x >> 5);
  int lane = threadIdx.x & 31, hi = lane >> 4, ln = lane & 15;
  int ct = wave & 7, pt = wave >> 3;
  int co0 = ct * 16;
  int p0 = pt * 16;
  int n = p0 / HWSZ, pp = p0 % HWSZ;
  int yy0 = pp / IMW, x0 = pp % IMW;
  v8f acc = zacc();
#pragma unroll
  for (int t = 0; t < 9; ++t) {
    int dy = t / 3 - 1, dx = t % 3 - 1;
    int ys = yy0 + dy;
    int xs = x0 + ln + dx;
    bool valid = (ys >= 0 && ys < IMH && xs >= 0 && xs < IMW);
    const _Float16* arow = w3h + ((size_t)(t * CCH) + co0 + ln) * CCH;
    const _Float16* brow = y2 + ((long)n * HWSZ + (long)ys * IMW + xs) * CCH;
    if (t < 8)  // gfx1250 global_prefetch_b8 on next tap's weights
      __builtin_prefetch(w3h + ((size_t)((t + 1) * CCH) + co0 + ln) * CCH, 0, 1);
#pragma unroll
    for (int k0 = 0; k0 < CCH; k0 += 32) {
      v16h b = valid ? frag_row_b(brow + k0, hi) : zfrag();
      acc = wmma32(frag_row_a(arow + k0, hi), b, acc);
    }
  }
#pragma unroll
  for (int r = 0; r < 8; ++r) {
    int co = co0 + hi * 8 + r;
    size_t idx = (size_t)(n * CCH + co) * HWSZ + (size_t)yy0 * IMW + x0 + ln;
    float v = acc[r] * s3[co] + b3[co] + xin[idx];
    out[idx] = v > 0.f ? v : 0.f;
  }
}

// ---------------- host launcher ----------------
extern "C" void kernel_launch(void* const* d_in, const int* in_sizes, int n_in,
                              void* d_out, int out_size, void* d_ws, size_t ws_size,
                              hipStream_t stream) {
  (void)in_sizes; (void)n_in; (void)out_size; (void)ws_size;
  const float* x        = (const float*)d_in[0];
  const float* conv1_w  = (const float*)d_in[1];
  const float* bn1_g    = (const float*)d_in[2];
  const float* bn1_b    = (const float*)d_in[3];
  const float* bn1_m    = (const float*)d_in[4];
  const float* bn1_v    = (const float*)d_in[5];
  const float* to_q     = (const float*)d_in[6];
  const float* to_kv    = (const float*)d_in[7];
  const float* to_out_w = (const float*)d_in[8];
  const float* to_out_b = (const float*)d_in[9];
  const float* rel_h    = (const float*)d_in[10];
  const float* rel_w    = (const float*)d_in[11];
  const float* conv3_w  = (const float*)d_in[12];
  const float* bn3_g    = (const float*)d_in[13];
  const float* bn3_b    = (const float*)d_in[14];
  const float* bn3_m    = (const float*)d_in[15];
  const float* bn3_v    = (const float*)d_in[16];
  float* out = (float*)d_out;
  char* ws = (char*)d_ws;

  size_t off = 0;
  auto carve = [&](size_t bytes) {
    size_t o = off; off = (off + bytes + 255) & ~(size_t)255; return o;
  };
  _Float16* xh    = (_Float16*)(ws + carve((size_t)NPIX * CCH * 2));
  _Float16* y1    = (_Float16*)(ws + carve((size_t)NPIX * CCH * 2));
  _Float16* y2    = (_Float16*)(ws + carve((size_t)NPIX * CCH * 2));
  _Float16* w1h   = (_Float16*)(ws + carve((size_t)CCH * CCH * 2));
  _Float16* tqh   = (_Float16*)(ws + carve((size_t)CCH * CCH * 2));
  _Float16* tkh   = (_Float16*)(ws + carve((size_t)CCH * CCH * 2));
  _Float16* tvh   = (_Float16*)(ws + carve((size_t)CCH * CCH * 2));
  _Float16* toh   = (_Float16*)(ws + carve((size_t)CCH * CCH * 2));
  _Float16* relwh = (_Float16*)(ws + carve(32 * 32 * 2));
  _Float16* relhh = (_Float16*)(ws + carve(32 * 32 * 2));
  _Float16* w3h   = (_Float16*)(ws + carve((size_t)9 * CCH * CCH * 2));
  float* s1 = (float*)(ws + carve(CCH * 4));
  float* b1 = (float*)(ws + carve(CCH * 4));
  float* s3 = (float*)(ws + carve(CCH * 4));
  float* b3 = (float*)(ws + carve(CCH * 4));

  prep_kernel<<<128, 256, 0, stream>>>(
      conv1_w, bn1_g, bn1_b, bn1_m, bn1_v, to_q, to_kv, to_out_w,
      rel_h, rel_w, conv3_w, bn3_g, bn3_b, bn3_m, bn3_v,
      w1h, tqh, tkh, tvh, toh, relwh, relhh, w3h, s1, b1, s3, b3);

  nchw_to_nhwc_f16<<<(NPIX * CCH) / 256, 256, 0, stream>>>(x, xh);

  // 8 cout-tiles x 4608 pixel-tiles, 1 tile per wave, 8 waves per block
  conv1x1_bn_relu<<<(8 * (NPIX / 16)) / 8, 256, 0, stream>>>(xh, w1h, s1, b1, y1);

  constexpr size_t SMEM =
      (size_t)(64 * CCH + 256 * CCH + CCH * LDV + 8 * 16 * LDP) * 2 +
      (size_t)(8 * 16 * 32) * 4 * 2 + (size_t)(64 * CCH) * 2;  // 266240 B
  halo_attn<<<NIMG * NBLK, 256, SMEM, stream>>>(y1, tqh, tkh, tvh, toh,
                                                relwh, relhh, to_out_b, y2);

  conv3x3_bn_add_relu<<<(8 * (NPIX / 16)) / 8, 256, 0, stream>>>(
      y2, w3h, s3, b3, x, out);
}
